// DIEN_24137716203863
// MI455X (gfx1250) — matrix operand order repaired and checked by reading, not compiled, not run on hardware
//
#include <hip/hip_runtime.h>
#include <hip/hip_bf16.h>

// ---------------------------------------------------------------------------
// DIEN forward for MI455X (gfx1250, wave32, WMMA 16x16x32 f16->f32).
// Requires ws_size >= ~62 MB.
// GRU/AUGRU scans: one wave32 per 16 batch rows, hidden state held in WMMA
// C/D-layout registers, weight B-fragments register-resident, and the
// data-dependent embedding gather software-pipelined two steps deep so its
// latency hides behind the per-step WMMA block. Gate nonlinearities lowered
// to v_exp_f32 + v_rcp_f32 (no IEEE div sequence on the recurrence path).
// ---------------------------------------------------------------------------

typedef __attribute__((ext_vector_type(16))) _Float16 v16h;
typedef __attribute__((ext_vector_type(8)))  _Float16 v8h;
typedef __attribute__((ext_vector_type(8)))  float    v8f;
typedef _Float16 h16;

#define B_    2048
#define T_    200
#define E_    32
#define NF_   2
#define SL_   8
#define DL_   16
#define D_    64
#define G3_   192   // 3*D
#define AIN_  256   // attention input width
#define DIN_  336   // dnn input width
#define DINP_ 352   // dnn input padded to 32
#define H1_   256
#define H2_   128

__device__ __forceinline__ v16h cat8(v8h lo, v8h hi) {
  return __builtin_shufflevector(lo, hi, 0,1,2,3,4,5,6,7,8,9,10,11,12,13,14,15);
}

// A fragment (16x32 f16): rows at base + row*stride, K window [k0, k0+32).
// Lane layout per ISA 7.12.2: lanes 0-15 -> K {0..7,16..23}; 16-31 -> {8..15,24..31}.
__device__ __forceinline__ v16h frag_a(const h16* base, int stride, int k0) {
  const int lane = threadIdx.x & 31;
  const int row  = lane & 15;
  const int ko   = (lane & 16) ? 8 : 0;
  const h16* p = base + (long long)row * stride + k0 + ko;
  v8h lo = *(const v8h*)(p);
  v8h hi = *(const v8h*)(p + 16);
  return cat8(lo, hi);
}

// B fragment (32x16): Wt stored [N][Kpad] f16 (i.e. B[k][n] = Wt[n][k]).
// Lane holds col n = ncol + (lane&15), k {0..7,16..23} (lanes 0-15) / {8..15,24..31} (16-31) from k0.
__device__ __forceinline__ v16h frag_b(const h16* Wt, int Kpad, int ncol, int k0) {
  const int lane = threadIdx.x & 31;
  const int n = ncol + (lane & 15);
  const int k = k0 + ((lane & 16) ? 8 : 0);
  const v8h* p = (const v8h*)(Wt + (long long)n * Kpad + k);
  return cat8(p[0], p[2]);
}

__device__ __forceinline__ v8f wmma_f16(v16h a, v16h b, v8f c) {
  return __builtin_amdgcn_wmma_f32_16x16x32_f16(false, a, false, b, (short)0, c,
                                                false, false);
}

__device__ __forceinline__ v8f vbcast(float s) {
  v8f c;
#pragma unroll
  for (int i = 0; i < 8; ++i) c[i] = s;
  return c;
}

// sigmoid via v_exp_f32 + v_rcp_f32 (no IEEE division sequence)
__device__ __forceinline__ float sigmoidf_(float x) {
  return __builtin_amdgcn_rcpf(1.f + __expf(-x));
}
// tanh(x) = 1 - 2/(exp(2x)+1); saturates correctly for |x| large
__device__ __forceinline__ float tanhf_(float x) {
  return 1.f - 2.f * __builtin_amdgcn_rcpf(__expf(2.f * x) + 1.f);
}

// ---------------------------------------------------------------------------
// weight prep kernels
// ---------------------------------------------------------------------------
__global__ void k_cvt(const float* __restrict__ src, h16* __restrict__ dst, int n) {
  int i = blockIdx.x * blockDim.x + threadIdx.x;
  if (i < n) dst[i] = (h16)src[i];
}

// src [K][N] f32 -> dst [N][Kpad] f16, zero-padded for k >= K
__global__ void k_cvt_t(const float* __restrict__ src, h16* __restrict__ dst,
                        int K, int N, int Kpad) {
  int i = blockIdx.x * blockDim.x + threadIdx.x;
  if (i >= N * Kpad) return;
  int n = i / Kpad, k = i % Kpad;
  dst[i] = (k < K) ? (h16)src[(long long)k * N + n] : (h16)0.f;
}

__global__ void k_target_emb(const int* __restrict__ tgt, const float* __restrict__ emb,
                             h16* __restrict__ out) {
  int i = blockIdx.x * blockDim.x + threadIdx.x;
  if (i >= B_ * D_) return;
  int b = i / D_, j = i % D_;
  int idx = tgt[b * NF_ + (j >> 5)];
  out[i] = (h16)emb[(long long)idx * E_ + (j & 31)];
}

// ---------------------------------------------------------------------------
// GRU interest-extractor scan
// ---------------------------------------------------------------------------
__global__ __launch_bounds__(32) void k_gru_scan(
    const int* __restrict__ hist_item, const int* __restrict__ lens,
    const float* __restrict__ emb,
    const h16* __restrict__ wih, const h16* __restrict__ whh,   // [192][64] f16
    const float* __restrict__ bih, const float* __restrict__ bhh,
    h16* __restrict__ interests)                                 // [B][T][64] f16
{
  __shared__ __align__(16) h16 xs[16 * D_];
  __shared__ __align__(16) h16 hs[16 * D_];
  const int lane = threadIdx.x;
  const int b0   = blockIdx.x * 16;
  const int nlo  = lane & 15;
  const int moff = (lane & 16) ? 8 : 0;

  int mylen[8];
#pragma unroll
  for (int r = 0; r < 8; ++r) mylen[r] = lens[b0 + moff + r];

  float brz[8], bin[4], bhn[4];
#pragma unroll
  for (int n = 0; n < 8; ++n) brz[n] = bih[n * 16 + nlo] + bhh[n * 16 + nlo];
#pragma unroll
  for (int n = 0; n < 4; ++n) {
    bin[n] = bih[(8 + n) * 16 + nlo];
    bhn[n] = bhh[(8 + n) * 16 + nlo];
  }

  v8f hfrag[4];
#pragma unroll
  for (int t4 = 0; t4 < 4; ++t4) hfrag[t4] = vbcast(0.f);
#pragma unroll
  for (int i = 0; i < 32; ++i) hs[i * 32 + lane] = (h16)0.f;

  // gather pipeline: lane -> (row, feature) of the 16x2 index matrix
  const int grow = lane >> 1, gfeat = lane & 1;
  const long long hist_base = (long long)(b0 + grow) * T_ * NF_ + gfeat;
  h16* xdst = xs + grow * D_ + gfeat * E_;

  // prologue: x_0 into LDS, index for t=1 in flight
  int idx_nxt = hist_item[hist_base + NF_];
  {
    int idx0 = hist_item[hist_base];
    const float4* src = (const float4*)(emb + (long long)idx0 * E_);
    float4 p[8];
#pragma unroll
    for (int i = 0; i < 8; ++i) p[i] = src[i];
#pragma unroll
    for (int i = 0; i < 8; ++i) {
      xdst[4 * i + 0] = (h16)p[i].x; xdst[4 * i + 1] = (h16)p[i].y;
      xdst[4 * i + 2] = (h16)p[i].z; xdst[4 * i + 3] = (h16)p[i].w;
    }
  }
  __syncthreads();

  for (int t = 0; t < T_; ++t) {
    // ---- prefetch x_{t+1} rows (address known) + index for t+2; these loads
    // ---- drain while the WMMA block below executes.
    float4 pref[8];
    {
      const float4* src = (const float4*)(emb + (long long)idx_nxt * E_);
#pragma unroll
      for (int i = 0; i < 8; ++i) pref[i] = src[i];
    }
    const int t2 = (t + 2 < T_) ? t + 2 : T_ - 1;
    const int idx_n2 = hist_item[hist_base + (long long)t2 * NF_];

    v16h ax0 = frag_a(xs, D_, 0);
    v16h ax1 = frag_a(xs, D_, 32);
    v16h ah0 = frag_a(hs, D_, 0);
    v16h ah1 = frag_a(hs, D_, 32);

    v8f acc_rz[8], acc_in[4], acc_hn[4];
#pragma unroll
    for (int n = 0; n < 8; ++n) {       // r,z gates: fuse x- and h-GEMMs
      v8f c = vbcast(brz[n]);
      c = wmma_f16(ax0, frag_b(wih, D_, n * 16, 0),  c);
      c = wmma_f16(ax1, frag_b(wih, D_, n * 16, 32), c);
      c = wmma_f16(ah0, frag_b(whh, D_, n * 16, 0),  c);
      c = wmma_f16(ah1, frag_b(whh, D_, n * 16, 32), c);
      acc_rz[n] = c;
    }
#pragma unroll
    for (int n = 0; n < 4; ++n) {       // n gate: i_n and h_n kept separate
      v8f ci = vbcast(bin[n]);
      ci = wmma_f16(ax0, frag_b(wih, D_, (8 + n) * 16, 0),  ci);
      ci = wmma_f16(ax1, frag_b(wih, D_, (8 + n) * 16, 32), ci);
      acc_in[n] = ci;
      v8f ch = vbcast(bhn[n]);
      ch = wmma_f16(ah0, frag_b(whh, D_, (8 + n) * 16, 0),  ch);
      ch = wmma_f16(ah1, frag_b(whh, D_, (8 + n) * 16, 32), ch);
      acc_hn[n] = ch;
    }
    __syncthreads();

#pragma unroll
    for (int t4 = 0; t4 < 4; ++t4) {
#pragma unroll
      for (int r = 0; r < 8; ++r) {
        float rg = sigmoidf_(acc_rz[t4][r]);
        float zg = sigmoidf_(acc_rz[t4 + 4][r]);
        float ng = tanhf_(acc_in[t4][r] + rg * acc_hn[t4][r]);
        float hold = hfrag[t4][r];
        float hnew = (1.f - zg) * ng + zg * hold;        // torch GRU convention
        const bool v = (t < mylen[r]);
        hnew = v ? hnew : hold;
        hfrag[t4][r] = hnew;
        hs[(moff + r) * D_ + t4 * 16 + nlo] = (h16)hnew;
        interests[((long long)(b0 + moff + r) * T_ + t) * D_ + t4 * 16 + nlo] =
            (h16)(v ? hnew : 0.f);
      }
    }

    // stage prefetched x_{t+1} into LDS (waits on the overlapped loads here)
#pragma unroll
    for (int i = 0; i < 8; ++i) {
      xdst[4 * i + 0] = (h16)pref[i].x; xdst[4 * i + 1] = (h16)pref[i].y;
      xdst[4 * i + 2] = (h16)pref[i].z; xdst[4 * i + 3] = (h16)pref[i].w;
    }
    idx_nxt = idx_n2;
    __syncthreads();
  }
}

// ---------------------------------------------------------------------------
// DIN attention MLP over all B*T rows: 256 -> 64 -> 16 -> 1, raw scores out.
// 4 independent waves per block, each owns a 16-row tile.
// ---------------------------------------------------------------------------
__global__ __launch_bounds__(128) void k_attention(
    const h16* __restrict__ interests, const h16* __restrict__ tgt,
    const h16* __restrict__ w1t /*[64][256]*/, const float* __restrict__ b1,
    const h16* __restrict__ w2t /*[16][64]*/,  const float* __restrict__ b2,
    const float* __restrict__ w3 /*[16]*/,     const float* __restrict__ b3,
    float* __restrict__ scores)
{
  __shared__ __align__(16) h16 ai[4][16 * AIN_];
  __shared__ __align__(16) h16 h1buf[4][16 * D_];
  __shared__ float h2[4][16 * 16];
  const int lane = threadIdx.x & 31;
  const int wv   = threadIdx.x >> 5;
  const long long g0 = ((long long)blockIdx.x * 4 + wv) * 16;
  const int nlo  = lane & 15;
  const int moff = (lane & 16) ? 8 : 0;
  h16* my_ai = ai[wv];

  { // build ai = [q, k, q-k, q*k]: lane -> (row, half of j-range)
    const int row = lane & 15;
    const int j0  = (lane & 16) ? 32 : 0;
    const long long g = g0 + row;
    const int b = (int)(g / T_);
    const h16* q = tgt + (long long)b * D_;
    const h16* k = interests + g * D_;
#pragma unroll
    for (int jj = 0; jj < 32; ++jj) {
      int j = j0 + jj;
      h16 qj = q[j], kj = k[j];
      my_ai[row * AIN_ + j]       = qj;
      my_ai[row * AIN_ + 64 + j]  = kj;
      my_ai[row * AIN_ + 128 + j] = (h16)(qj - kj);
      my_ai[row * AIN_ + 192 + j] = (h16)(qj * kj);
    }
  }
  __syncthreads();

  v8f a1[4];
#pragma unroll
  for (int n = 0; n < 4; ++n) {
    v8f c = vbcast(b1[n * 16 + nlo]);
#pragma unroll
    for (int kt = 0; kt < 8; ++kt)
      c = wmma_f16(frag_a(my_ai, AIN_, kt * 32), frag_b(w1t, AIN_, n * 16, kt * 32), c);
    a1[n] = c;
  }
  h16* my_h1 = h1buf[wv];
#pragma unroll
  for (int n = 0; n < 4; ++n)
#pragma unroll
    for (int r = 0; r < 8; ++r)
      my_h1[(moff + r) * D_ + n * 16 + nlo] = (h16)fmaxf(a1[n][r], 0.f);
  __syncthreads();

  v8f c2 = vbcast(b2[nlo]);
  c2 = wmma_f16(frag_a(my_h1, D_, 0),  frag_b(w2t, D_, 0, 0),  c2);
  c2 = wmma_f16(frag_a(my_h1, D_, 32), frag_b(w2t, D_, 0, 32), c2);
  float* my_h2 = h2[wv];
#pragma unroll
  for (int r = 0; r < 8; ++r)
    my_h2[(moff + r) * 16 + nlo] = fmaxf(c2[r], 0.f);
  __syncthreads();

  if (lane < 16) {
    float s = b3[0];
#pragma unroll
    for (int j = 0; j < 16; ++j) s += my_h2[lane * 16 + j] * w3[j];
    scores[g0 + lane] = s;
  }
}

// masked softmax over T per batch row, in place
__global__ __launch_bounds__(256) void k_softmax(float* __restrict__ scores,
                                                 const int* __restrict__ lens) {
  __shared__ float red[256];
  const int b = blockIdx.x, tid = threadIdx.x;
  const int len = lens[b];
  const bool act = tid < T_;
  float val = 0.f;
  if (act) {
    val = scores[(long long)b * T_ + tid];
    if (tid >= len) val = -1e9f;
  }
  red[tid] = act ? val : -3.0e38f;
  __syncthreads();
  for (int st = 128; st > 0; st >>= 1) {
    if (tid < st) red[tid] = fmaxf(red[tid], red[tid + st]);
    __syncthreads();
  }
  const float mx = red[0];
  __syncthreads();
  const float e = act ? __expf(val - mx) : 0.f;
  red[tid] = e;
  __syncthreads();
  for (int st = 128; st > 0; st >>= 1) {
    if (tid < st) red[tid] += red[tid + st];
    __syncthreads();
  }
  const float inv = __builtin_amdgcn_rcpf(red[0]);
  if (act) scores[(long long)b * T_ + tid] = e * inv;
}

// ---------------------------------------------------------------------------
// AUGRU interest-evolving scan: z gate scaled by attention, keep last hidden.
// x-fragments (interests) software-pipelined one step ahead.
// ---------------------------------------------------------------------------
__global__ __launch_bounds__(32) void k_augru_scan(
    const h16* __restrict__ interests, const float* __restrict__ att,
    const int* __restrict__ lens,
    const h16* __restrict__ wih, const h16* __restrict__ whh,
    const float* __restrict__ bih, const float* __restrict__ bhh,
    float* __restrict__ hlast)                                   // [B][64] f32
{
  __shared__ __align__(16) h16 hs[16 * D_];
  __shared__ float as[16];
  const int lane = threadIdx.x;
  const int b0   = blockIdx.x * 16;
  const int nlo  = lane & 15;
  const int moff = (lane & 16) ? 8 : 0;

  int mylen[8];
#pragma unroll
  for (int r = 0; r < 8; ++r) mylen[r] = lens[b0 + moff + r];

  float brz[8], bin[4], bhn[4];
#pragma unroll
  for (int n = 0; n < 8; ++n) brz[n] = bih[n * 16 + nlo] + bhh[n * 16 + nlo];
#pragma unroll
  for (int n = 0; n < 4; ++n) {
    bin[n] = bih[(8 + n) * 16 + nlo];
    bhn[n] = bhh[(8 + n) * 16 + nlo];
  }

  v8f hfrag[4];
#pragma unroll
  for (int t4 = 0; t4 < 4; ++t4) hfrag[t4] = vbcast(0.f);
#pragma unroll
  for (int i = 0; i < 32; ++i) hs[i * 32 + lane] = (h16)0.f;
  __syncthreads();

  // prologue: x-fragments for t=0 in flight
  const h16* xb0 = interests + (long long)b0 * T_ * D_;
  v16h axn0 = frag_a(xb0, T_ * D_, 0);
  v16h axn1 = frag_a(xb0, T_ * D_, 32);

  for (int t = 0; t < T_; ++t) {
    v16h ax0 = axn0, ax1 = axn1;
    { // prefetch x-fragments for t+1; overlap with WMMA block
      const int tn = (t + 1 < T_) ? t + 1 : T_ - 1;
      const h16* xbn = interests + ((long long)b0 * T_ + tn) * D_;
      axn0 = frag_a(xbn, T_ * D_, 0);
      axn1 = frag_a(xbn, T_ * D_, 32);
    }
    if (lane < 16) as[lane] = att[(long long)(b0 + lane) * T_ + t];
    __syncthreads();

    v16h ah0 = frag_a(hs, D_, 0);
    v16h ah1 = frag_a(hs, D_, 32);

    v8f acc_rz[8], acc_in[4], acc_hn[4];
#pragma unroll
    for (int n = 0; n < 8; ++n) {
      v8f c = vbcast(brz[n]);
      c = wmma_f16(ax0, frag_b(wih, D_, n * 16, 0),  c);
      c = wmma_f16(ax1, frag_b(wih, D_, n * 16, 32), c);
      c = wmma_f16(ah0, frag_b(whh, D_, n * 16, 0),  c);
      c = wmma_f16(ah1, frag_b(whh, D_, n * 16, 32), c);
      acc_rz[n] = c;
    }
#pragma unroll
    for (int n = 0; n < 4; ++n) {
      v8f ci = vbcast(bin[n]);
      ci = wmma_f16(ax0, frag_b(wih, D_, (8 + n) * 16, 0),  ci);
      ci = wmma_f16(ax1, frag_b(wih, D_, (8 + n) * 16, 32), ci);
      acc_in[n] = ci;
      v8f ch = vbcast(bhn[n]);
      ch = wmma_f16(ah0, frag_b(whh, D_, (8 + n) * 16, 0),  ch);
      ch = wmma_f16(ah1, frag_b(whh, D_, (8 + n) * 16, 32), ch);
      acc_hn[n] = ch;
    }
    __syncthreads();

#pragma unroll
    for (int t4 = 0; t4 < 4; ++t4) {
#pragma unroll
      for (int r = 0; r < 8; ++r) {
        const float a = as[moff + r];
        float rg = sigmoidf_(acc_rz[t4][r]);
        float zg = a * sigmoidf_(acc_rz[t4 + 4][r]);     // attentional update gate
        float ng = tanhf_(acc_in[t4][r] + rg * acc_hn[t4][r]);
        float hold = hfrag[t4][r];
        float hnew = (1.f - zg) * hold + zg * ng;        // AUGRU convention
        hnew = (t < mylen[r]) ? hnew : hold;
        hfrag[t4][r] = hnew;
        hs[(moff + r) * D_ + t4 * 16 + nlo] = (h16)hnew;
      }
    }
    __syncthreads();
  }
#pragma unroll
  for (int t4 = 0; t4 < 4; ++t4)
#pragma unroll
    for (int r = 0; r < 8; ++r)
      hlast[(long long)(b0 + moff + r) * D_ + t4 * 16 + nlo] = hfrag[t4][r];
}

// ---------------------------------------------------------------------------
// DNN head
// ---------------------------------------------------------------------------
__global__ void k_build_dnnin(const int* __restrict__ sparse, const float* __restrict__ emb,
                              const float* __restrict__ dense, const float* __restrict__ hlast,
                              float* __restrict__ dnnin) {
  int i = blockIdx.x * blockDim.x + threadIdx.x;
  if (i >= B_ * DIN_) return;
  int b = i / DIN_, c = i % DIN_;
  float v;
  if (c < E_ * SL_) {
    int idx = sparse[b * SL_ + (c >> 5)];
    v = emb[(long long)idx * E_ + (c & 31)];
  } else if (c < E_ * SL_ + DL_) {
    v = dense[b * DL_ + (c - E_ * SL_)];
  } else {
    v = hlast[(long long)b * D_ + (c - E_ * SL_ - DL_)];
  }
  dnnin[i] = v;
}

__global__ __launch_bounds__(256) void k_bnstats(const float* __restrict__ dnnin,
                                                 float* __restrict__ mu,
                                                 float* __restrict__ rsig) {
  __shared__ float s1[256], s2[256];
  const int c = blockIdx.x, tid = threadIdx.x;
  float a = 0.f, b = 0.f;
  for (int r = tid; r < B_; r += 256) {
    float x = dnnin[(long long)r * DIN_ + c];
    a += x; b += x * x;
  }
  s1[tid] = a; s2[tid] = b;
  __syncthreads();
  for (int st = 128; st > 0; st >>= 1) {
    if (tid < st) { s1[tid] += s1[tid + st]; s2[tid] += s2[tid + st]; }
    __syncthreads();
  }
  if (tid == 0) {
    float m = s1[0] / (float)B_;
    float var = s2[0] / (float)B_ - m * m;
    mu[c] = m;
    rsig[c] = rsqrtf(var + 1e-5f);
  }
}

__global__ void k_bnnorm(const float* __restrict__ dnnin, const float* __restrict__ mu,
                         const float* __restrict__ rsig, const float* __restrict__ gamma,
                         const float* __restrict__ beta, h16* __restrict__ x1) {
  int i = blockIdx.x * blockDim.x + threadIdx.x;
  if (i >= B_ * DINP_) return;
  int b = i / DINP_, c = i % DINP_;
  float v = 0.f;
  if (c < DIN_)
    v = (dnnin[(long long)b * DIN_ + c] - mu[c]) * rsig[c] * gamma[c] + beta[c];
  x1[i] = (h16)v;
}

// Y[M,N] = relu(X @ Wt^T + bias), f16 output. X row-major [M][Kpad] f16.
__global__ __launch_bounds__(128) void k_gemm_relu_h16(
    const h16* __restrict__ X, int Kpad, int Kt,
    const h16* __restrict__ Wt, const float* __restrict__ bias,
    int N, h16* __restrict__ Y) {
  const int lane = threadIdx.x & 31, wv = threadIdx.x >> 5;
  const int m0 = (blockIdx.x * 4 + wv) * 16;
  const int nlo = lane & 15, moff = (lane & 16) ? 8 : 0;
  const h16* Xrow = X + (long long)m0 * Kpad;
  for (int nt = 0; nt < N / 16; ++nt) {
    v8f c = vbcast(bias[nt * 16 + nlo]);
    for (int kt = 0; kt < Kt; ++kt)
      c = wmma_f16(frag_a(Xrow, Kpad, kt * 32), frag_b(Wt, Kpad, nt * 16, kt * 32), c);
#pragma unroll
    for (int r = 0; r < 8; ++r)
      Y[(long long)(m0 + moff + r) * N + nt * 16 + nlo] = (h16)fmaxf(c[r], 0.f);
  }
}

// same, f32 output
__global__ __launch_bounds__(128) void k_gemm_relu_f32(
    const h16* __restrict__ X, int Kpad, int Kt,
    const h16* __restrict__ Wt, const float* __restrict__ bias,
    int N, float* __restrict__ Y) {
  const int lane = threadIdx.x & 31, wv = threadIdx.x >> 5;
  const int m0 = (blockIdx.x * 4 + wv) * 16;
  const int nlo = lane & 15, moff = (lane & 16) ? 8 : 0;
  const h16* Xrow = X + (long long)m0 * Kpad;
  for (int nt = 0; nt < N / 16; ++nt) {
    v8f c = vbcast(bias[nt * 16 + nlo]);
    for (int kt = 0; kt < Kt; ++kt)
      c = wmma_f16(frag_a(Xrow, Kpad, kt * 32), frag_b(Wt, Kpad, nt * 16, kt * 32), c);
#pragma unroll
    for (int r = 0; r < 8; ++r)
      Y[(long long)(m0 + moff + r) * N + nt * 16 + nlo] = fmaxf(c[r], 0.f);
  }
}

__global__ void k_final(const float* __restrict__ x2, const float* __restrict__ ow,
                        const float* __restrict__ ob, float* __restrict__ out) {
  int b = blockIdx.x * blockDim.x + threadIdx.x;
  if (b >= B_) return;
  float s = ob[0];
  for (int j = 0; j < H2_; ++j) s += x2[(long long)b * H2_ + j] * ow[j];
  out[b] = s;
}

// ---------------------------------------------------------------------------
extern "C" void kernel_launch(void* const* d_in, const int* in_sizes, int n_in,
                              void* d_out, int out_size, void* d_ws, size_t ws_size,
                              hipStream_t stream) {
  const int*   target_item = (const int*)d_in[0];
  const int*   hist_item   = (const int*)d_in[1];
  const int*   lens        = (const int*)d_in[3];
  const float* dense       = (const float*)d_in[4];
  const int*   sparse      = (const int*)d_in[5];
  const float* emb         = (const float*)d_in[6];
  const float* gru_wih  = (const float*)d_in[7];
  const float* gru_whh  = (const float*)d_in[8];
  const float* gru_bih  = (const float*)d_in[9];
  const float* gru_bhh  = (const float*)d_in[10];
  const float* aug_wih  = (const float*)d_in[11];
  const float* aug_whh  = (const float*)d_in[12];
  const float* aug_bih  = (const float*)d_in[13];
  const float* aug_bhh  = (const float*)d_in[14];
  const float* att_w1   = (const float*)d_in[15];
  const float* att_b1   = (const float*)d_in[16];
  const float* att_w2   = (const float*)d_in[17];
  const float* att_b2   = (const float*)d_in[18];
  const float* att_w3   = (const float*)d_in[19];
  const float* att_b3   = (const float*)d_in[20];
  const float* bn_gamma = (const float*)d_in[21];
  const float* bn_beta  = (const float*)d_in[22];
  const float* dnn_w1   = (const float*)d_in[23];
  const float* dnn_b1   = (const float*)d_in[24];
  const float* dnn_w2   = (const float*)d_in[25];
  const float* dnn_b2   = (const float*)d_in[26];
  const float* out_w    = (const float*)d_in[27];
  const float* out_b    = (const float*)d_in[28];
  float* out = (float*)d_out;

  char* wsp = (char*)d_ws;
  auto alloc = [&](size_t bytes) -> char* {
    char* p = wsp;
    wsp += (bytes + 255) & ~(size_t)255;
    return p;
  };
  h16* wihG   = (h16*)alloc((size_t)G3_ * D_ * 2);
  h16* whhG   = (h16*)alloc((size_t)G3_ * D_ * 2);
  h16* wihA   = (h16*)alloc((size_t)G3_ * D_ * 2);
  h16* whhA   = (h16*)alloc((size_t)G3_ * D_ * 2);
  h16* w1t    = (h16*)alloc((size_t)64 * AIN_ * 2);
  h16* w2t    = (h16*)alloc((size_t)16 * D_ * 2);
  h16* dw1t   = (h16*)alloc((size_t)H1_ * DINP_ * 2);
  h16* dw2t   = (h16*)alloc((size_t)H2_ * H1_ * 2);
  h16* tgt16  = (h16*)alloc((size_t)B_ * D_ * 2);
  h16* inter  = (h16*)alloc((size_t)B_ * T_ * D_ * 2);     // 52.4 MB
  float* sc   = (float*)alloc((size_t)B_ * T_ * 4);
  float* hl   = (float*)alloc((size_t)B_ * D_ * 4);
  float* din  = (float*)alloc((size_t)B_ * DIN_ * 4);
  float* mu   = (float*)alloc((size_t)DIN_ * 4);
  float* rsig = (float*)alloc((size_t)DIN_ * 4);
  h16* x1     = (h16*)alloc((size_t)B_ * DINP_ * 2);
  h16* y1     = (h16*)alloc((size_t)B_ * H1_ * 2);
  float* x2   = (float*)alloc((size_t)B_ * H2_ * 4);

  // --- weight prep (f16 copies / transposes; [N][K] layouts stay as-is) ---
  k_cvt<<<(G3_ * D_ + 255) / 256, 256, 0, stream>>>(gru_wih, wihG, G3_ * D_);
  k_cvt<<<(G3_ * D_ + 255) / 256, 256, 0, stream>>>(gru_whh, whhG, G3_ * D_);
  k_cvt<<<(G3_ * D_ + 255) / 256, 256, 0, stream>>>(aug_wih, wihA, G3_ * D_);
  k_cvt<<<(G3_ * D_ + 255) / 256, 256, 0, stream>>>(aug_whh, whhA, G3_ * D_);
  k_cvt_t<<<(64 * AIN_ + 255) / 256, 256, 0, stream>>>(att_w1, w1t, AIN_, 64, AIN_);
  k_cvt_t<<<(16 * D_ + 255) / 256, 256, 0, stream>>>(att_w2, w2t, D_, 16, D_);
  k_cvt_t<<<(H1_ * DINP_ + 255) / 256, 256, 0, stream>>>(dnn_w1, dw1t, DIN_, H1_, DINP_);
  k_cvt_t<<<(H2_ * H1_ + 255) / 256, 256, 0, stream>>>(dnn_w2, dw2t, H1_, H2_, H1_);
  k_target_emb<<<(B_ * D_ + 255) / 256, 256, 0, stream>>>(target_item, emb, tgt16);

  // --- interest extractor GRU ---
  k_gru_scan<<<B_ / 16, 32, 0, stream>>>(hist_item, lens, emb, wihG, whhG,
                                         gru_bih, gru_bhh, inter);
  // --- DIN attention + softmax ---
  k_attention<<<(B_ * T_) / 64, 128, 0, stream>>>(inter, tgt16, w1t, att_b1,
                                                  w2t, att_b2, att_w3, att_b3, sc);
  k_softmax<<<B_, 256, 0, stream>>>(sc, lens);
  // --- interest evolving AUGRU ---
  k_augru_scan<<<B_ / 16, 32, 0, stream>>>(inter, sc, lens, wihA, whhA,
                                           aug_bih, aug_bhh, hl);
  // --- DNN head ---
  k_build_dnnin<<<(B_ * DIN_ + 255) / 256, 256, 0, stream>>>(sparse, emb, dense, hl, din);
  k_bnstats<<<DIN_, 256, 0, stream>>>(din, mu, rsig);
  k_bnnorm<<<(B_ * DINP_ + 255) / 256, 256, 0, stream>>>(din, mu, rsig, bn_gamma,
                                                         bn_beta, x1);
  k_gemm_relu_h16<<<(B_ / 16) / 4, 128, 0, stream>>>(x1, DINP_, DINP_ / 32, dw1t,
                                                     dnn_b1, H1_, y1);
  k_gemm_relu_f32<<<(B_ / 16) / 4, 128, 0, stream>>>(y1, H1_, H1_ / 32, dw2t,
                                                     dnn_b2, H2_, x2);
  k_final<<<(B_ + 255) / 256, 256, 0, stream>>>(x2, out_w, out_b, out);
}
